// MoE_LoRA_CLIP_80530636800252
// MI455X (gfx1250) — matrix operand order repaired and checked by reading, not compiled, not run on hardware
//
#include <hip/hip_runtime.h>
#include <hip/hip_bf16.h>

// ---------------------------------------------------------------------------
// MoE-LoRA MLP forward on MI455X (gfx1250), bf16 WMMA path.
//   N=16384 tokens, D=1024, F=4096, E=8, R=32, SCALING=0.5
// Pipeline:
//   K0 router+pack:   logits/softmax/argmax, routing & expert_choice out,
//                     bf16(x) -> A1[:, 0:1024]
//   K1 weight packs:  B1[F x 1280] = [fc1_w | down_B_w], B2[D x 4352] =
//                     [fc2_w | up_B_w], dA[256 x 1024], uA[256 x 4096]  (bf16)
//   K2 WMMA gemm<LORA>: A1[:, :1024] x dA^T -> masked 0.5*(h+A_b) -> A1[:,1024:]
//   K3 WMMA gemm<GELU>: A1 x B1^T (K=1280) + fc1_b + 0.5*down_B_b[e] -> gelu
//                       -> bf16 A2[:, 0:4096]
//   K4 WMMA gemm<LORA>: A2[:, :4096] x uA^T -> masked -> A2[:, 4096:]
//   K5 WMMA gemm<OUT>:  A2 x B2^T (K=4352) + fc2_b + 0.5*up_B_b[e] -> out f32
// ---------------------------------------------------------------------------

#define NT 16384   // B*S tokens
#define DD 1024
#define FF 4096
#define EE 8
#define RR 32
#define K1DIM 1280  // D + E*R
#define K2DIM 4352  // F + E*R

typedef __attribute__((ext_vector_type(8)))  float  v8f;
typedef __attribute__((ext_vector_type(8)))  __bf16 v8bf;
typedef __attribute__((ext_vector_type(16))) __bf16 v16bf;

__device__ __forceinline__ unsigned short f2bf(float f) {
  unsigned int u = __float_as_uint(f);
  unsigned int r = u + 0x7FFFu + ((u >> 16) & 1u);   // round-to-nearest-even
  return (unsigned short)(r >> 16);
}

__device__ __forceinline__ v16bf cat16(v8bf lo, v8bf hi) {
  return __builtin_shufflevector(lo, hi, 0,1,2,3,4,5,6,7,8,9,10,11,12,13,14,15);
}

// A-matrix fragment (16x32 bf16): lane holds row M=lane%16.
// elements 0..7  <- K = k0 + (lane/16)*8 .. +7
// elements 8..15 <- K = k0 + 16 + (lane/16)*8 .. +7
__device__ __forceinline__ v16bf load_fragA(const unsigned short* base, int lda,
                                            int row, int k0, int g) {
  const __bf16* p = reinterpret_cast<const __bf16*>(base) + (size_t)row * lda + k0 + g * 8;
  v8bf lo = *reinterpret_cast<const v8bf*>(p);
  v8bf hi = *reinterpret_cast<const v8bf*>(p + 16);
  return cat16(lo, hi);
}

// B-matrix fragment (32x16 bf16), B stored as [Ncols x K] (K contiguous):
// lane holds col N=lane%16; elements 0..7 <- K = k0 + (lane/16)*8 .. +7, 8..15 <- K = k0 + 16 + (lane/16)*8 .. +7
__device__ __forceinline__ v16bf load_fragB(const unsigned short* base, int ldb,
                                            int col, int k0, int g) {
  const __bf16* p = reinterpret_cast<const __bf16*>(base) + (size_t)col * ldb + k0 + g * 8;
  v8bf lo = *reinterpret_cast<const v8bf*>(p);
  v8bf hi = *reinterpret_cast<const v8bf*>(p + 16);
  return cat16(lo, hi);
}

// ---------------------------------------------------------------------------
// K0: router + bf16(x) pack. One wave (32 lanes) per token.
// ---------------------------------------------------------------------------
__launch_bounds__(256)
__global__ void router_pack_kernel(const float* __restrict__ x,
                                   const float* __restrict__ rw,
                                   const float* __restrict__ rb,
                                   unsigned short* __restrict__ A1,
                                   float* __restrict__ routing,
                                   float* __restrict__ ec,
                                   int* __restrict__ idx) {
  const int lane = threadIdx.x & 31;
  const int wave = threadIdx.x >> 5;
  const int t = blockIdx.x * 8 + wave;
  const float* xr = x + (size_t)t * DD;
  unsigned short* a1r = A1 + (size_t)t * K1DIM;

  float acc[EE];
#pragma unroll
  for (int e = 0; e < EE; ++e) acc[e] = 0.0f;

#pragma unroll
  for (int c = 0; c < 8; ++c) {
    const int d = c * 128 + lane * 4;
    float4 xv = *reinterpret_cast<const float4*>(xr + d);
    ushort4 s;
    s.x = f2bf(xv.x); s.y = f2bf(xv.y); s.z = f2bf(xv.z); s.w = f2bf(xv.w);
    *reinterpret_cast<ushort4*>(a1r + d) = s;
#pragma unroll
    for (int e = 0; e < EE; ++e) {
      float4 wv = *reinterpret_cast<const float4*>(rw + (size_t)e * DD + d);
      acc[e] += xv.x * wv.x + xv.y * wv.y + xv.z * wv.z + xv.w * wv.w;
    }
  }
#pragma unroll
  for (int e = 0; e < EE; ++e)
    for (int o = 16; o > 0; o >>= 1) acc[e] += __shfl_xor(acc[e], o, 32);

  if (lane == 0) {
    float lg[EE], mx = -3.4e38f;
#pragma unroll
    for (int e = 0; e < EE; ++e) { lg[e] = acc[e] + rb[e]; mx = fmaxf(mx, lg[e]); }
    float p[EE], s = 0.0f;
#pragma unroll
    for (int e = 0; e < EE; ++e) { p[e] = __expf(lg[e] - mx); s += p[e]; }
    const float inv = 1.0f / s;
    int best = 0; float bv = lg[0];
#pragma unroll
    for (int e = 1; e < EE; ++e) if (lg[e] > bv) { bv = lg[e]; best = e; }
    idx[t] = best;
#pragma unroll
    for (int e = 0; e < EE; ++e) {
      const float pr = p[e] * inv;
      routing[(size_t)t * EE + e] = pr;
      const float hard = (e == best) ? 1.0f : 0.0f;
      ec[(size_t)t * EE + e] = (hard - pr) + pr;   // match reference fp order
    }
  }
}

// ---------------------------------------------------------------------------
// K1: weight packs (fp32 -> bf16, with K-concatenation of LoRA B weights)
// ---------------------------------------------------------------------------
__global__ void pack_B1(const float* __restrict__ fc1w, const float* __restrict__ dBw,
                        unsigned short* __restrict__ B1) {
  const int tid = blockIdx.x * 256 + threadIdx.x;        // F*K1DIM threads
  const int f = tid / K1DIM, k = tid % K1DIM;
  float v;
  if (k < DD) v = fc1w[(size_t)f * DD + k];
  else { const int q = k - DD, e = q >> 5, r = q & 31;
         v = dBw[((size_t)e * FF + f) * RR + r]; }
  B1[tid] = f2bf(v);
}

__global__ void pack_B2(const float* __restrict__ fc2w, const float* __restrict__ uBw,
                        unsigned short* __restrict__ B2) {
  const int tid = blockIdx.x * 256 + threadIdx.x;        // D*K2DIM threads
  const int d = tid / K2DIM, k = tid % K2DIM;
  float v;
  if (k < FF) v = fc2w[(size_t)d * FF + k];
  else { const int q = k - FF, e = q >> 5, r = q & 31;
         v = uBw[((size_t)e * DD + d) * RR + r]; }
  B2[tid] = f2bf(v);
}

__global__ void convert_bf16(const float* __restrict__ src, unsigned short* __restrict__ dst) {
  const int tid = blockIdx.x * 256 + threadIdx.x;
  dst[tid] = f2bf(src[tid]);
}

// ---------------------------------------------------------------------------
// WMMA GEMM: C[M x N] = A[M x K] * B[N x K]^T, bf16 in / f32 accum.
// Block 128 threads = 4 waves (2x2), wave tile 64x64, block tile 128x128.
// MODE 0 (LORA): out bf16 = (col_expert==idx[row]) ? 0.5*(acc + bias[col]) : 0
// MODE 1 (GELU): out bf16 = gelu(acc + bias[col] + 0.5*ebias[e*ebld+col])
// MODE 2 (OUT):  out f32  =      acc + bias[col] + 0.5*ebias[e*ebld+col]
// ---------------------------------------------------------------------------
template <int MODE>
__launch_bounds__(128)
__global__ void gemm_bf16(const unsigned short* __restrict__ A,
                          const unsigned short* __restrict__ B,
                          int K, int lda, int ldb,
                          const int* __restrict__ idx,
                          const float* __restrict__ bias,
                          const float* __restrict__ ebias,
                          unsigned short* __restrict__ obf,
                          float* __restrict__ ofp,
                          int ldo, int ebld) {
  const int lane = threadIdx.x & 31;
  const int wave = threadIdx.x >> 5;
  const int g = lane >> 4;
  const int ln = lane & 15;
  const int rowBase = blockIdx.y * 128 + (wave >> 1) * 64;
  const int colBase = blockIdx.x * 128 + (wave & 1) * 64;

  v8f acc[4][4];
#pragma unroll
  for (int i = 0; i < 4; ++i)
#pragma unroll
    for (int j = 0; j < 4; ++j) acc[i][j] = (v8f)0.0f;

  for (int k0 = 0; k0 < K; k0 += 32) {
    v16bf af[4], bf[4];
#pragma unroll
    for (int i = 0; i < 4; ++i) af[i] = load_fragA(A, lda, rowBase + i * 16 + ln, k0, g);
#pragma unroll
    for (int j = 0; j < 4; ++j) bf[j] = load_fragB(B, ldb, colBase + j * 16 + ln, k0, g);
#pragma unroll
    for (int i = 0; i < 4; ++i)
#pragma unroll
      for (int j = 0; j < 4; ++j)
        acc[i][j] = __builtin_amdgcn_wmma_f32_16x16x32_bf16(
            false, af[i], false, bf[j], (short)0, acc[i][j], false, false);
  }

#pragma unroll
  for (int i = 0; i < 4; ++i) {
#pragma unroll
    for (int d = 0; d < 8; ++d) {
      const int row = rowBase + i * 16 + d + 8 * g;
      const int e_t = idx[row];
#pragma unroll
      for (int j = 0; j < 4; ++j) {
        const int c = colBase + j * 16 + ln;
        const float v = acc[i][j][d];
        if constexpr (MODE == 0) {
          const float o = ((c >> 5) == e_t) ? 0.5f * (v + bias[c]) : 0.0f;
          obf[(size_t)row * ldo + c] = f2bf(o);
        } else if constexpr (MODE == 1) {
          const float t = v + bias[c] + 0.5f * ebias[(size_t)e_t * ebld + c];
          const float gl = 0.5f * t * (1.0f + erff(t * 0.70710678118654752f));
          obf[(size_t)row * ldo + c] = f2bf(gl);
        } else {
          ofp[(size_t)row * ldo + c] = v + bias[c] + 0.5f * ebias[(size_t)e_t * ebld + c];
        }
      }
    }
  }
}

// ---------------------------------------------------------------------------
extern "C" void kernel_launch(void* const* d_in, const int* in_sizes, int n_in,
                              void* d_out, int out_size, void* d_ws, size_t ws_size,
                              hipStream_t stream) {
  const float* x    = (const float*)d_in[0];
  const float* rw   = (const float*)d_in[1];
  const float* rb   = (const float*)d_in[2];
  const float* fc1w = (const float*)d_in[3];
  const float* fc1b = (const float*)d_in[4];
  const float* fc2w = (const float*)d_in[5];
  const float* fc2b = (const float*)d_in[6];
  const float* dAw  = (const float*)d_in[7];
  const float* dAb  = (const float*)d_in[8];
  const float* dBw  = (const float*)d_in[9];
  const float* dBb  = (const float*)d_in[10];
  const float* uAw  = (const float*)d_in[11];
  const float* uAb  = (const float*)d_in[12];
  const float* uBw  = (const float*)d_in[13];
  const float* uBb  = (const float*)d_in[14];

  // workspace carving (all sizes 256B multiples)
  char* ws = (char*)d_ws;
  constexpr size_t szA1 = (size_t)NT * K1DIM * 2;          // 41,943,040
  constexpr size_t szA2 = (size_t)NT * K2DIM * 2;          // 142,606,336
  constexpr size_t szB1 = (size_t)FF * K1DIM * 2;          // 10,485,760
  constexpr size_t szB2 = (size_t)DD * K2DIM * 2;          //  8,912,896
  constexpr size_t szdA = (size_t)EE * RR * DD * 2;        //     524,288
  constexpr size_t szuA = (size_t)EE * RR * FF * 2;        //   2,097,152
  unsigned short* A1 = (unsigned short*)(ws);
  unsigned short* A2 = (unsigned short*)(ws + szA1);
  unsigned short* B1 = (unsigned short*)(ws + szA1 + szA2);
  unsigned short* B2 = (unsigned short*)(ws + szA1 + szA2 + szB1);
  unsigned short* dA = (unsigned short*)(ws + szA1 + szA2 + szB1 + szB2);
  unsigned short* uA = (unsigned short*)(ws + szA1 + szA2 + szB1 + szB2 + szdA);
  int*            idx = (int*)(ws + szA1 + szA2 + szB1 + szB2 + szdA + szuA);

  float* out     = (float*)d_out;
  float* routing = out + (size_t)NT * DD;
  float* ec      = routing + (size_t)NT * EE;

  // K0: router + bf16(x)
  router_pack_kernel<<<NT / 8, 256, 0, stream>>>(x, rw, rb, A1, routing, ec, idx);
  // K1: weight packs
  convert_bf16<<<(EE * RR * DD) / 256, 256, 0, stream>>>(dAw, dA);
  convert_bf16<<<(EE * RR * FF) / 256, 256, 0, stream>>>(uAw, uA);
  pack_B1<<<(FF * K1DIM) / 256, 256, 0, stream>>>(fc1w, dBw, B1);
  pack_B2<<<(DD * K2DIM) / 256, 256, 0, stream>>>(fc2w, uBw, B2);
  // K2: down-LoRA rank activations -> A1[:, 1024:1280]
  gemm_bf16<0><<<dim3(2, NT / 128), 128, 0, stream>>>(
      A1, dA, DD, K1DIM, DD, idx, dAb, nullptr, A1 + DD, nullptr, K1DIM, 0);
  // K3: h1 = [x|h_dn] x [fc1|down_B]^T, +biases, gelu -> bf16 A2[:, 0:4096]
  gemm_bf16<1><<<dim3(FF / 128, NT / 128), 128, 0, stream>>>(
      A1, B1, K1DIM, K1DIM, K1DIM, idx, fc1b, dBb, A2, nullptr, K2DIM, FF);
  // K4: up-LoRA rank activations -> A2[:, 4096:4352]
  gemm_bf16<0><<<dim3(2, NT / 128), 128, 0, stream>>>(
      A2, uA, FF, K2DIM, FF, idx, uAb, nullptr, A2 + FF, nullptr, K2DIM, 0);
  // K5: out = [a|h_up] x [fc2|up_B]^T + biases -> f32 out
  gemm_bf16<2><<<dim3(DD / 128, NT / 128), 128, 0, stream>>>(
      A2, B2, K2DIM, K2DIM, K2DIM, idx, fc2b, uBb, nullptr, out, DD, DD);
}
